// DAttention_29652454211863
// MI455X (gfx1250) — matrix operand order repaired and checked by reading, not compiled, not run on hardware
//
#include <hip/hip_runtime.h>
#include <hip/hip_bf16.h>
#include <math.h>

typedef __attribute__((ext_vector_type(16))) _Float16 v16h;
typedef __attribute__((ext_vector_type(8)))  float    v8f;
typedef __attribute__((ext_vector_type(4)))  int      v4i;

#define Bsz 2
#define Cch 384
#define HW 1024
#define NH 12
#define NG 6
#define HC 32
#define GC 64
#define NS 1024
#define RPE_W 63
#define SCALE 0.17677669529663687f  // 32^-0.5

// ---------------------------------------------------------------------------
// gfx1250 async global->LDS copy (ASYNCcnt path), with sync fallback if the
// builtins are absent on this toolchain. Probe round 2 showed the builtin's
// params are int4 pointers in AS1 (global) / AS3 (LDS).
// ---------------------------------------------------------------------------
#if defined(__has_builtin)
#if __has_builtin(__builtin_amdgcn_global_load_async_to_lds_b128) && \
    __has_builtin(__builtin_amdgcn_s_wait_asynccnt)
#define HAVE_ASYNC_LDS 1
#endif
#endif
#ifndef HAVE_ASYNC_LDS
#define HAVE_ASYNC_LDS 0
#endif

__device__ __forceinline__ void cp_b128_to_lds(const float* g, float* l) {
#if HAVE_ASYNC_LDS
  typedef __attribute__((address_space(1))) v4i gv4_t;
  typedef __attribute__((address_space(3))) v4i lv4_t;
  __builtin_amdgcn_global_load_async_to_lds_b128((gv4_t*)g, (lv4_t*)l, 0, 0);
#else
  *(float4*)l = *(const float4*)g;
#endif
}
__device__ __forceinline__ void async_copies_done() {
#if HAVE_ASYNC_LDS
  __builtin_amdgcn_s_wait_asynccnt(0);
#endif
}

// ---------------------------------------------------------------------------
// WMMA GEMM: Y[b,m,n] = sum_k W[m,k] * X[b,k,n] + bias[m]
// W: [M,K] f32 row-major, X: [B,K,N] f32, Y: [B,M,N] f32.
// block = 128 threads = 4 waves; each wave computes 16x32 (two 16x16 tiles,
// shared A fragment); block covers 16 rows x 128 cols. K fully unrolled.
// grid.x = (M/16)*(N/128), grid.y = batch
// ---------------------------------------------------------------------------
template <int M, int N, int K>
__global__ __launch_bounds__(128)
void gemm_wmma_kernel(const float* __restrict__ W, const float* __restrict__ bias,
                      const float* __restrict__ X, float* __restrict__ Y) {
  constexpr int NB = N / 128;
  int lane = threadIdx.x & 31;
  int wave = threadIdx.x >> 5;
  int mt   = blockIdx.x / NB;
  int nt   = blockIdx.x % NB;
  int m16  = mt * 16;
  int n0   = nt * 128 + wave * 32;
  int b    = blockIdx.y;
  X += (size_t)b * K * N;
  Y += (size_t)b * M * N;

  int lm    = lane & 15;
  int abase = (lane < 16) ? 0 : 8;    // unused; A is loaded from kbase (k = 16h + s)
  int kbase = (lane < 16) ? 0 : 16;   // B-fragment K base
  int rofs  = (lane < 16) ? 0 : 8;    // C/D row offset
  int mrow  = m16 + lm;
  int ncol  = n0 + lm;

  v8f acc0 = {}, acc1 = {};
#pragma unroll
  for (int k0 = 0; k0 < K; k0 += 32) {
    v16h a, b0, b1;
    const float* wr = W + (size_t)mrow * K + k0 + kbase;
#pragma unroll
    for (int i = 0; i < 8; ++i) a[i]     = (_Float16)wr[i];
#pragma unroll
    for (int i = 0; i < 8; ++i) a[8 + i] = (_Float16)wr[8 + i];
    const float* xc = X + (size_t)(k0 + kbase) * N + ncol;
    if (k0 + 32 < K) __builtin_prefetch(xc + (size_t)32 * N, 0, 1);
#pragma unroll
    for (int i = 0; i < 16; ++i) {
      b0[i] = (_Float16)xc[(size_t)i * N];
      b1[i] = (_Float16)xc[(size_t)i * N + 16];
    }
    acc0 = __builtin_amdgcn_wmma_f32_16x16x32_f16(false, a, false, b0, (short)0, acc0, false, false);
    acc1 = __builtin_amdgcn_wmma_f32_16x16x32_f16(false, a, false, b1, (short)0, acc1, false, false);
  }
#pragma unroll
  for (int r = 0; r < 8; ++r) {
    int row = m16 + rofs + r;
    float bs = bias[row];
    Y[(size_t)row * N + ncol]      = acc0[r] + bs;
    Y[(size_t)row * N + ncol + 16] = acc1[r] + bs;
  }
}

// ---------------------------------------------------------------------------
// Offset branch: depthwise 7x7 conv + LN(ch) + GELU + 1x1 -> tanh -> pos(y,x)
// one thread per (bg, pixel); 64-ch vector staged in LDS.
// ---------------------------------------------------------------------------
__global__ __launch_bounds__(128)
void offset_kernel(const float* __restrict__ q, const float* __restrict__ dw_w,
                   const float* __restrict__ dw_b, const float* __restrict__ ln_g,
                   const float* __restrict__ ln_b, const float* __restrict__ pw_w,
                   float* __restrict__ pos) {
  __shared__ float obuf[128][GC + 1];
  int tid = blockIdx.x * 128 + threadIdx.x;   // bg*1024 + p  (12*1024 total)
  int p  = tid & 1023;
  int bg = tid >> 10;
  int b  = bg / NG, g = bg % NG;
  int y  = p >> 5, x = p & 31;
  const float* qg = q + ((size_t)b * Cch + g * GC) * HW;
  float* o = obuf[threadIdx.x];

  for (int c = 0; c < GC; ++c) {
    float acc = dw_b[c];
    const float* wc = dw_w + c * 49;
    const float* qc = qg + (size_t)c * HW;
    for (int dy = 0; dy < 7; ++dy) {
      int yy = y + dy - 3;
      if (yy < 0 || yy > 31) continue;
      for (int dx = 0; dx < 7; ++dx) {
        int xx = x + dx - 3;
        if (xx < 0 || xx > 31) continue;
        acc += qc[yy * 32 + xx] * wc[dy * 7 + dx];
      }
    }
    o[c] = acc;
  }
  float mu = 0.f;
  for (int c = 0; c < GC; ++c) mu += o[c];
  mu *= (1.f / GC);
  float var = 0.f;
  for (int c = 0; c < GC; ++c) { float d = o[c] - mu; var += d * d; }
  var *= (1.f / GC);
  float inv = rsqrtf(var + 1e-5f);
  float oy = 0.f, ox = 0.f;
  for (int c = 0; c < GC; ++c) {
    float t = (o[c] - mu) * inv * ln_g[c] + ln_b[c];
    t = 0.5f * t * (1.f + erff(t * 0.70710678118654752f));  // exact GELU
    oy += t * pw_w[c];
    ox += t * pw_w[GC + c];
  }
  float off_y = tanhf(oy) * (2.0f / 32.0f);   // OFF_FACTOR / Hs
  float off_x = tanhf(ox) * (2.0f / 32.0f);
  float ref_y = ((y + 0.5f) * (1.f / 32.f)) * 2.f - 1.f;
  float ref_x = ((x + 0.5f) * (1.f / 32.f)) * 2.f - 1.f;
  pos[(size_t)bg * 2048 + p * 2 + 0] = off_y + ref_y;
  pos[(size_t)bg * 2048 + p * 2 + 1] = off_x + ref_x;
}

// ---------------------------------------------------------------------------
// grid_sample of x at pos (align_corners=true, zero padding)
// one thread per (bg, c, s); total 12*64*1024
// ---------------------------------------------------------------------------
__global__ __launch_bounds__(256)
void sample_kernel(const float* __restrict__ x, const float* __restrict__ pos,
                   float* __restrict__ xs) {
  int tid = blockIdx.x * 256 + threadIdx.x;
  int s  = tid & 1023;
  int c  = (tid >> 10) & 63;
  int bg = tid >> 16;
  int b  = bg / NG, g = bg % NG;
  float2 pp = *(const float2*)(pos + (size_t)bg * 2048 + s * 2);
  float gx = (pp.y + 1.f) * 0.5f * 31.f;
  float gy = (pp.x + 1.f) * 0.5f * 31.f;
  float x0f = floorf(gx), y0f = floorf(gy);
  float wx = gx - x0f, wy = gy - y0f;
  int x0 = (int)x0f, y0 = (int)y0f;
  const float* img = x + ((size_t)b * Cch + g * GC + c) * HW;
  float r = 0.f;
#pragma unroll
  for (int j = 0; j < 2; ++j)
#pragma unroll
    for (int i = 0; i < 2; ++i) {
      int xi = x0 + i, yi = y0 + j;
      float w = (i ? wx : 1.f - wx) * (j ? wy : 1.f - wy);
      if (xi >= 0 && xi <= 31 && yi >= 0 && yi <= 31)
        r += img[yi * 32 + xi] * w;
    }
  xs[((size_t)b * Cch + g * GC + c) * NS + s] = r;
}

__device__ __forceinline__ float bilin_rpe(const float* __restrict__ img,
                                           float gx, float gy) {
  float x0f = floorf(gx), y0f = floorf(gy);
  float wx = gx - x0f, wy = gy - y0f;
  int x0 = (int)x0f, y0 = (int)y0f;
  float acc = 0.f;
#pragma unroll
  for (int j = 0; j < 2; ++j)
#pragma unroll
    for (int i = 0; i < 2; ++i) {
      int xi = x0 + i, yi = y0 + j;
      float w = (i ? wx : 1.f - wx) * (j ? wy : 1.f - wy);
      if (xi >= 0 && xi <= RPE_W - 1 && yi >= 0 && yi <= RPE_W - 1)
        acc += img[yi * RPE_W + xi] * w;
    }
  return acc;
}

// ---------------------------------------------------------------------------
// Fused attention (flash style): S = q^T k * SCALE + RPE bias; online softmax;
// O += P * v^T.  One wave owns 16 queries; block = 4 waves = 64 queries.
// K/V 32x32 tiles are staged once per block into LDS via async global->LDS
// copies (shared by all 4 waves). grid = 24 heads * 16 query-blocks.
// ---------------------------------------------------------------------------
__global__ __launch_bounds__(128)
void attn_kernel(const float* __restrict__ q, const float* __restrict__ kbuf,
                 const float* __restrict__ vbuf, const float* __restrict__ pos,
                 const float* __restrict__ rpe, float* __restrict__ out) {
  __shared__ float    Kt[HC][36];         // [channel][key], padded rows (16B)
  __shared__ float    Vt[HC][36];
  __shared__ _Float16 Plds[4][16][32];
  int lane = threadIdx.x & 31;
  int wave = threadIdx.x >> 5;
  int blk  = blockIdx.x;
  int mblk = blk & 15;
  int bh   = blk >> 4;        // 0..23
  int b = bh / NH, h = bh % NH;
  int bg = b * NG + (h >> 1);
  int m0 = mblk * 64 + wave * 16;

  const float* qh = q    + ((size_t)b * Cch + h * HC) * HW;
  const float* kh = kbuf + ((size_t)b * Cch + h * HC) * NS;
  const float* vh = vbuf + ((size_t)b * Cch + h * HC) * NS;
  const float* rp = rpe + (size_t)h * RPE_W * RPE_W;
  const float* ps = pos + (size_t)bg * 2048;

  int lm    = lane & 15;
  int abase = (lane < 16) ? 0 : 8;
  int kbase = (lane < 16) ? 0 : 16;
  int rofs  = (lane < 16) ? 0 : 8;

  // A fragment of q^T : A[m,k] = qh[k*HW + m]  (loaded once)
  v16h Aq;
  {
    int m = m0 + lm;
#pragma unroll
    for (int i = 0; i < 8; ++i) Aq[i]     = (_Float16)qh[(size_t)(kbase + i) * HW + m];
#pragma unroll
    for (int i = 0; i < 8; ++i) Aq[8 + i] = (_Float16)qh[(size_t)(kbase + 8 + i) * HW + m];
  }

  v8f O0 = {}, O1 = {};
  float mrun[8], lrun[8];
#pragma unroll
  for (int r = 0; r < 8; ++r) { mrun[r] = -1e30f; lrun[r] = 0.f; }

  for (int n0 = 0; n0 < NS; n0 += 32) {
    __syncthreads();  // previous iteration's LDS consumers are done
    // cooperative staging: K tile + V tile, 256 float4 each, 128 threads
    {
      int t = threadIdx.x;
#pragma unroll
      for (int i = 0; i < 2; ++i) {
        int f = t + i * 128;                 // 0..255
        int ch = f >> 3, c4 = (f & 7) * 4;
        cp_b128_to_lds(kh + (size_t)ch * NS + n0 + c4, &Kt[ch][c4]);
      }
#pragma unroll
      for (int i = 0; i < 2; ++i) {
        int f = t + i * 128;
        int ch = f >> 3, c4 = (f & 7) * 4;
        cp_b128_to_lds(vh + (size_t)ch * NS + n0 + c4, &Vt[ch][c4]);
      }
    }
    async_copies_done();   // per-wave drain of ASYNCcnt before the barrier
    __syncthreads();

    // K fragments from LDS: B[k=channel, n=key]
    v16h Bk0, Bk1;
#pragma unroll
    for (int i = 0; i < 16; ++i) {
      Bk0[i] = (_Float16)Kt[kbase + i][lm];
      Bk1[i] = (_Float16)Kt[kbase + i][16 + lm];
    }
    v8f z = {};
    v8f S0 = __builtin_amdgcn_wmma_f32_16x16x32_f16(false, Aq, false, Bk0, (short)0, z, false, false);
    v8f S1 = __builtin_amdgcn_wmma_f32_16x16x32_f16(false, Aq, false, Bk1, (short)0, z, false, false);

    // scale + RPE bias (bilinear lookup at (q_ref - pos)*0.5, align_corners)
    for (int t = 0; t < 2; ++t) {
      int n = n0 + t * 16 + lm;
      float2 pp = *(const float2*)(ps + 2 * n);
      float py = pp.x, px = pp.y;
      v8f& S = t ? S1 : S0;
#pragma unroll
      for (int r = 0; r < 8; ++r) {
        int m = m0 + rofs + r;
        int my = m >> 5, mx = m & 31;
        float qy = ((my + 0.5f) * (1.f / 32.f)) * 2.f - 1.f;
        float qx = ((mx + 0.5f) * (1.f / 32.f)) * 2.f - 1.f;
        float dy = (qy - py) * 0.5f;
        float dx = (qx - px) * 0.5f;
        float gx = (dx + 1.f) * 0.5f * (RPE_W - 1);
        float gy = (dy + 1.f) * 0.5f * (RPE_W - 1);
        S[r] = S[r] * SCALE + bilin_rpe(rp, gx, gy);
      }
    }

    // online softmax: row reductions within 16-lane halves
#pragma unroll
    for (int r = 0; r < 8; ++r) {
      float mx = fmaxf(S0[r], S1[r]);
      mx = fmaxf(mx, __shfl_xor(mx, 1));
      mx = fmaxf(mx, __shfl_xor(mx, 2));
      mx = fmaxf(mx, __shfl_xor(mx, 4));
      mx = fmaxf(mx, __shfl_xor(mx, 8));
      float mnew  = fmaxf(mrun[r], mx);
      float alpha = __expf(mrun[r] - mnew);
      mrun[r] = mnew;
      float p0 = __expf(S0[r] - mnew);
      float p1 = __expf(S1[r] - mnew);
      S0[r] = p0; S1[r] = p1;
      float rs = p0 + p1;
      rs += __shfl_xor(rs, 1);
      rs += __shfl_xor(rs, 2);
      rs += __shfl_xor(rs, 4);
      rs += __shfl_xor(rs, 8);
      lrun[r] = lrun[r] * alpha + rs;
      O0[r] *= alpha;
      O1[r] *= alpha;
    }

    // repack P (C layout) -> A fragment via per-wave LDS tile
    __syncthreads();
#pragma unroll
    for (int r = 0; r < 8; ++r) {
      Plds[wave][rofs + r][lm]      = (_Float16)S0[r];
      Plds[wave][rofs + r][16 + lm] = (_Float16)S1[r];
    }
    __syncthreads();
    v16h Ap;
    {
      const _Float16* row = &Plds[wave][lm][0];
#pragma unroll
      for (int i = 0; i < 8; ++i) Ap[i]     = row[kbase + i];
#pragma unroll
      for (int i = 0; i < 8; ++i) Ap[8 + i] = row[kbase + 8 + i];
    }

    // V fragments from LDS: B[k=key, n=channel] = Vt[channel][key]
    v16h Bv0, Bv1;
#pragma unroll
    for (int i = 0; i < 16; ++i) {
      Bv0[i] = (_Float16)Vt[lm][kbase + i];
      Bv1[i] = (_Float16)Vt[16 + lm][kbase + i];
    }
    O0 = __builtin_amdgcn_wmma_f32_16x16x32_f16(false, Ap, false, Bv0, (short)0, O0, false, false);
    O1 = __builtin_amdgcn_wmma_f32_16x16x32_f16(false, Ap, false, Bv1, (short)0, O1, false, false);
  }

  // normalize and store transposed back to [b, h*32+c, m]
  float* ob = out + (size_t)b * Cch * HW;
#pragma unroll
  for (int r = 0; r < 8; ++r) {
    float inv = 1.f / lrun[r];
    int m = m0 + rofs + r;
    ob[(size_t)(h * HC + lm) * HW + m]      = O0[r] * inv;
    ob[(size_t)(h * HC + 16 + lm) * HW + m] = O1[r] * inv;
  }
}

// ---------------------------------------------------------------------------
extern "C" void kernel_launch(void* const* d_in, const int* in_sizes, int n_in,
                              void* d_out, int out_size, void* d_ws, size_t ws_size,
                              hipStream_t stream) {
  (void)in_sizes; (void)n_in; (void)out_size; (void)ws_size;
  const float* x     = (const float*)d_in[0];
  const float* Wq    = (const float*)d_in[1];
  const float* bq    = (const float*)d_in[2];
  const float* Wk    = (const float*)d_in[3];
  const float* bk    = (const float*)d_in[4];
  const float* Wv    = (const float*)d_in[5];
  const float* bv    = (const float*)d_in[6];
  const float* Wo    = (const float*)d_in[7];
  const float* bo    = (const float*)d_in[8];
  const float* dw_w  = (const float*)d_in[9];
  const float* dw_b  = (const float*)d_in[10];
  const float* ln_g  = (const float*)d_in[11];
  const float* ln_b  = (const float*)d_in[12];
  const float* pw_w  = (const float*)d_in[13];
  const float* rpe   = (const float*)d_in[14];
  float* y = (float*)d_out;

  const size_t TEN = (size_t)Bsz * Cch * HW;   // 786432
  float* q   = (float*)d_ws;
  float* pos = q + TEN;                        // 12*1024*2 = 24576
  float* xs  = pos + 24576;
  float* kk  = xs + TEN;
  float* vv  = kk + TEN;
  float* ao  = vv + TEN;

  dim3 ggrid((Cch / 16) * (HW / 128), Bsz);    // 24*8 = 192 blocks, batch in y

  // q = Wq x + bq
  gemm_wmma_kernel<Cch, HW, Cch><<<ggrid, 128, 0, stream>>>(Wq, bq, x, q);
  // deformable offsets -> sampling positions
  offset_kernel<<<(NG * Bsz * HW) / 128, 128, 0, stream>>>(q, dw_w, dw_b, ln_g, ln_b, pw_w, pos);
  // grid_sample x at pos
  sample_kernel<<<(NG * Bsz * GC * NS) / 256, 256, 0, stream>>>(x, pos, xs);
  // k, v projections of sampled features
  gemm_wmma_kernel<Cch, NS, Cch><<<ggrid, 128, 0, stream>>>(Wk, bk, xs, kk);
  gemm_wmma_kernel<Cch, NS, Cch><<<ggrid, 128, 0, stream>>>(Wv, bv, xs, vv);
  // fused attention with RPE bias + online softmax
  attn_kernel<<<Bsz * NH * (HW / 64), 128, 0, stream>>>(q, kk, vv, pos, rpe, ao);
  // output projection
  gemm_wmma_kernel<Cch, HW, Cch><<<ggrid, 128, 0, stream>>>(Wo, bo, ao, y);
}
